// VRAE_LSTM_61899068670616
// MI455X (gfx1250) — compile-verified, matrix operand order checked by reading, not run on hardware
//
#include <hip/hip_runtime.h>
#include <hip/hip_bf16.h>

typedef __attribute__((ext_vector_type(16))) _Float16 v16h;
typedef __attribute__((ext_vector_type(8)))  float    v8f;
typedef __attribute__((ext_vector_type(4)))  unsigned int v4u;
typedef __attribute__((ext_vector_type(8)))  int      v8i;
typedef __attribute__((ext_vector_type(4)))  int      v4i;

#define TSTEPS 1024
#define BATCH  128
#define HDIM   256
#define FDIM   64
#define NWG    4

#if defined(__has_builtin)
#  if __has_builtin(__builtin_amdgcn_tensor_load_to_lds)
#    define HAVE_TDM 1
#  else
#    define HAVE_TDM 0
#  endif
#else
#  define HAVE_TDM 0
#endif

// ---------------------------------------------------------------- helpers

__device__ __forceinline__ v8f splat8(float v) {
  v8f r;
#pragma unroll
  for (int i = 0; i < 8; ++i) r[i] = v;
  return r;
}

__device__ __forceinline__ v8f wmma16(v16h a, v16h b, v8f c) {
  // D = A(16x32 f16) * B(32x16 f16) + C(16x16 f32)
  return __builtin_amdgcn_wmma_f32_16x16x32_f16(
      /*neg_a=*/false, a, /*neg_b=*/false, b,
      /*c_mod=*/(short)0, c, /*reuse_a=*/false, /*reuse_b=*/false);
}

__device__ __forceinline__ float sigm(float x) {
  return 1.0f / (1.0f + __expf(-x));
}
__device__ __forceinline__ float tanh_f(float x) {
  x = fminf(fmaxf(x, -15.0f), 15.0f);
  float e = __expf(2.0f * x);
  return (e - 1.0f) / (e + 1.0f);
}

// Packed A-fragment layout for activation matrices [B=128, H=256]:
//   hbuf[mt(8)][kc(8)][lane(32)][e(16)]  (f16)
// 16-bit A 16x32 tile: lane = r%16 + 16*((k/8)&1), e = k%8 + 8*(k/16)
__device__ __forceinline__ int hpack_idx(int r, int j) {
  int mt = r >> 4, rr = r & 15, kc = j >> 5, ko = j & 31;
  int ln = rr + (((ko >> 3) & 1) << 4);
  int e  = (ko & 7) + ((ko >> 4) << 3);
  return ((mt * 8 + kc) * 32 + ln) * 16 + e;
}

// Build an A fragment of x_t ([128,64] fp32, row pitch T*F) on the fly
__device__ __forceinline__ v16h ld_xfrag(const float* __restrict__ x, int t,
                                         int mt, int kc, int lane) {
  int rr = lane & 15, hb = lane >> 4;
  const float* p = x + ((size_t)(mt * 16 + rr) * TSTEPS + t) * FDIM + kc * 32 + hb * 8;
  float4 u0 = *(const float4*)(p);
  float4 u1 = *(const float4*)(p + 4);
  float4 u2 = *(const float4*)(p + 16);
  float4 u3 = *(const float4*)(p + 20);
  v16h a;
  a[0]=(_Float16)u0.x; a[1]=(_Float16)u0.y; a[2]=(_Float16)u0.z; a[3]=(_Float16)u0.w;
  a[4]=(_Float16)u1.x; a[5]=(_Float16)u1.y; a[6]=(_Float16)u1.z; a[7]=(_Float16)u1.w;
  a[8]=(_Float16)u2.x; a[9]=(_Float16)u2.y; a[10]=(_Float16)u2.z; a[11]=(_Float16)u2.w;
  a[12]=(_Float16)u3.x; a[13]=(_Float16)u3.y; a[14]=(_Float16)u3.z; a[15]=(_Float16)u3.w;
  return a;
}

__device__ __forceinline__ void grid_barrier(unsigned* ctr, unsigned target) {
  __threadfence();
  __syncthreads();
  if (threadIdx.x == 0) {
    __hip_atomic_fetch_add(ctr, 1u, __ATOMIC_ACQ_REL, __HIP_MEMORY_SCOPE_AGENT);
    while (__hip_atomic_load(ctr, __ATOMIC_ACQUIRE, __HIP_MEMORY_SCOPE_AGENT) < target)
      __builtin_amdgcn_s_sleep(1);
  }
  __syncthreads();
}

#if HAVE_TDM
// Tensor Data Mover: 1-row tile of n8 8-byte elements, global -> LDS.
// D# per CDNA5 ISA ch.8: group0 = {count/type/lds_addr/global_addr},
// group1 = {data_size, tensor_dim0/1, tile_dim0/1, strides}.
__device__ __forceinline__ void tdm_load_1d(unsigned lds_off,
                                            const void* gptr,
                                            unsigned nbytes) {
  unsigned n8 = nbytes >> 3;  // 8-byte elements
  unsigned long long ga = (unsigned long long)(uintptr_t)gptr;
  v4u g0;
  g0[0] = 1u;                                             // count=1 (valid user D#)
  g0[1] = lds_off;                                        // lds_addr [63:32]
  g0[2] = (unsigned)(ga & 0xFFFFFFFFu);                   // global_addr[31:0]
  g0[3] = (unsigned)((ga >> 32) & 0x1FFFFFFu) | (2u << 30); // addr[56:32], type=2
  v8i g1;
  g1[0] = (int)(3u << 16);                  // workgroup_mask=0, data_size=3 (8B)
  g1[1] = (int)((n8 & 0xFFFFu) << 16);      // tensor_dim0[15:0] @ bits 63:48
  g1[2] = (int)((n8 >> 16) & 0xFFFFu) | (1 << 16); // tensor_dim0[31:16], tensor_dim1=1
  g1[3] = (int)((n8 & 0xFFFFu) << 16);      // tensor_dim1 hi=0, tile_dim0 @ 127:112
  g1[4] = 1;                                // tile_dim1=1, tile_dim2=0
  g1[5] = (int)n8;                          // tensor_dim0_stride[31:0]
  g1[6] = 0;                                // stride hi, tensor_dim1_stride lo
  g1[7] = 0;
  v4i g2 = {0, 0, 0, 0};
  v4i g3 = {0, 0, 0, 0};
#if __clang_major__ >= 23
  v8i g4 = {0, 0, 0, 0, 0, 0, 0, 0};
  __builtin_amdgcn_tensor_load_to_lds(g0, g1, g2, g3, g4, 0);
#else
  __builtin_amdgcn_tensor_load_to_lds(g0, g1, g2, g3, 0);
#endif
}
#endif  // HAVE_TDM

// ---------------------------------------------------------------- packing

__global__ void zero_kernel(float* p, int n) {
  int i = blockIdx.x * blockDim.x + threadIdx.x;
  if (i < n) p[i] = 0.0f;
}

// Pack W[G,K] (fp32, used as h@W.T) into WMMA B fragments (f16):
//   frag[(nt*Kc + kc)*32 + lane][e], B_tile[k][n] = W[nt*16+n][kc*32+k]
//   lane = n + 16*((k/8)&1), e = k%8 + 8*(k/16)
// If W2 != nullptr, packs W + W2 (decoder weight folding).
__global__ void pack_bfrags(const float* __restrict__ W, const float* __restrict__ W2,
                            int G, int K, _Float16* __restrict__ dst) {
  int t = blockIdx.x * blockDim.x + threadIdx.x;
  int Kc = K >> 5;
  int total = (G >> 4) * Kc * 32;
  if (t >= total) return;
  int lane = t & 31, tile = t >> 5;
  int nt = tile / Kc, kc = tile % Kc;
  int n = lane & 15, hb = lane >> 4;
  _Float16* d = dst + (size_t)t * 16;
#pragma unroll
  for (int e = 0; e < 16; ++e) {
    int k = ((e >> 3) << 4) + (hb << 3) + (e & 7);
    size_t idx = (size_t)(nt * 16 + n) * K + kc * 32 + k;
    float v = W[idx];
    if (W2) v += W2[idx];
    d[e] = (_Float16)v;
  }
}

// ---------------------------------------------------------------- encoder

__global__ __launch_bounds__(256) void enc_lstm_kernel(
    const float* __restrict__ x, const float* __restrict__ bvec,
    const _Float16* __restrict__ gwih, const _Float16* __restrict__ gwhh,
    _Float16* __restrict__ hbuf, unsigned* __restrict__ bar) {
  extern __shared__ _Float16 lds[];
  _Float16* lwih = lds;                  // 16 tiles * 2 kc * 512
  _Float16* lwhh = lds + 16 * 2 * 512;   // 16 tiles * 8 kc * 512
  const int tid = threadIdx.x, wg = blockIdx.x;
  const int lane = tid & 31, wave = tid >> 5;

  // Stage this WG's weight slice (gate cols {g*256 + wg*64 .. +64}) into LDS
#if HAVE_TDM
  if (wave == 0) {
    for (int u = 0; u < 16; ++u) {
      int g = u >> 2, tt = u & 3, nt = g * 16 + wg * 4 + tt;
      tdm_load_1d((unsigned)(uintptr_t)(void*)(lwih + u * 2 * 512),
                  (const void*)(gwih + (size_t)nt * 2 * 512), 2 * 512 * 2);
      tdm_load_1d((unsigned)(uintptr_t)(void*)(lwhh + u * 8 * 512),
                  (const void*)(gwhh + (size_t)nt * 8 * 512), 8 * 512 * 2);
    }
    __builtin_amdgcn_s_wait_tensorcnt(0);
  }
#else
  for (int u = 0; u < 16; ++u) {
    int g = u >> 2, tt = u & 3, nt = g * 16 + wg * 4 + tt;
    const uint4* s1 = (const uint4*)(gwih + (size_t)nt * 2 * 512);
    uint4* d1 = (uint4*)(lwih + u * 2 * 512);
    for (int i = tid; i < 128; i += 256) d1[i] = s1[i];
    const uint4* s2 = (const uint4*)(gwhh + (size_t)nt * 8 * 512);
    uint4* d2 = (uint4*)(lwhh + u * 8 * 512);
    for (int i = tid; i < 512; i += 256) d2[i] = s2[i];
  }
#endif
  __syncthreads();

  const int hct = wave >> 1;           // h-col-tile within WG slice (0..3)
  const int mt0 = (wave & 1) * 4;      // first row-tile of this wave
  const int n = lane & 15;
  const int jcol = wg * 64 + hct * 16 + n;

  float bg[4];
#pragma unroll
  for (int g = 0; g < 4; ++g) bg[g] = bvec[g * HDIM + jcol];

  v8f cst[4];
#pragma unroll
  for (int q = 0; q < 4; ++q) cst[q] = splat8(0.0f);

#pragma unroll 1
  for (int t = 0; t < TSTEPS; ++t) {
    const _Float16* hsrc = hbuf + (size_t)(t & 1) * 32768;
    _Float16*       hdst = hbuf + (size_t)((t + 1) & 1) * 32768;
#pragma unroll 1
    for (int q = 0; q < 4; ++q) {
      int mt = mt0 + q;
      v16h ah[8];
#pragma unroll
      for (int kc = 0; kc < 8; ++kc)
        ah[kc] = *(const v16h*)(hsrc + ((mt * 8 + kc) * 32 + lane) * 16);
      v16h ax0 = ld_xfrag(x, t, mt, 0, lane);
      v16h ax1 = ld_xfrag(x, t, mt, 1, lane);
      if (t + 1 < TSTEPS)
        __builtin_prefetch(x + ((size_t)(mt * 16 + n) * TSTEPS + (t + 1)) * FDIM, 0, 1);

      v8f acc[4];
#pragma unroll
      for (int g = 0; g < 4; ++g) acc[g] = splat8(bg[g]);
#pragma unroll
      for (int kc = 0; kc < 8; ++kc) {
#pragma unroll
        for (int g = 0; g < 4; ++g) {
          int u = g * 4 + hct;
          v16h bf = *(const v16h*)(lwhh + ((u * 8 + kc) * 32 + lane) * 16);
          acc[g] = wmma16(ah[kc], bf, acc[g]);
        }
      }
#pragma unroll
      for (int kc = 0; kc < 2; ++kc) {
        v16h a = kc ? ax1 : ax0;
#pragma unroll
        for (int g = 0; g < 4; ++g) {
          int u = g * 4 + hct;
          v16h bf = *(const v16h*)(lwih + ((u * 2 + kc) * 32 + lane) * 16);
          acc[g] = wmma16(a, bf, acc[g]);
        }
      }
#pragma unroll
      for (int k = 0; k < 8; ++k) {
        float iv = sigm(acc[0][k]);
        float fv = sigm(acc[1][k]);
        float gv = tanh_f(acc[2][k]);
        float ov = sigm(acc[3][k]);
        float cn = fv * cst[q][k] + iv * gv;
        cst[q][k] = cn;
        float hn = ov * tanh_f(cn);
        int r = mt * 16 + k + ((lane >> 4) << 3);
        hdst[hpack_idx(r, jcol)] = (_Float16)hn;
      }
    }
    grid_barrier(bar, (unsigned)(NWG * (t + 1)));
  }
}

// ------------------------------------------------- posterior + reparam

__global__ __launch_bounds__(256) void posterior_kernel(
    const float* __restrict__ eps,
    const float* __restrict__ bmp, const float* __restrict__ blp,
    const _Float16* __restrict__ gwmp, const _Float16* __restrict__ gwlp,
    const _Float16* __restrict__ hTa, _Float16* __restrict__ z0buf,
    float* __restrict__ omu, float* __restrict__ ols) {
  const int tid = threadIdx.x;
  const int lane = tid & 31, wave = tid >> 5;
  const int n = lane & 15;
  const int mt = wave;  // 8 row-tiles, one per wave
  v16h ah[8];
#pragma unroll
  for (int kc = 0; kc < 8; ++kc)
    ah[kc] = *(const v16h*)(hTa + ((mt * 8 + kc) * 32 + lane) * 16);
#pragma unroll 1
  for (int s = 0; s < 16; ++s) {
    int nt = s, j = nt * 16 + n;
    v8f am = splat8(bmp[j]);
    v8f al = splat8(blp[j]);
#pragma unroll
    for (int kc = 0; kc < 8; ++kc) {
      v16h bm = *(const v16h*)(gwmp + ((nt * 8 + kc) * 32 + lane) * 16);
      v16h bl = *(const v16h*)(gwlp + ((nt * 8 + kc) * 32 + lane) * 16);
      am = wmma16(ah[kc], bm, am);
      al = wmma16(ah[kc], bl, al);
    }
#pragma unroll
    for (int k = 0; k < 8; ++k) {
      int r = mt * 16 + k + ((lane >> 4) << 3);
      int idx = r * HDIM + j;
      float m = am[k], l = al[k];
      omu[idx] = m;
      ols[idx] = l;
      float z = eps[idx] * __expf(0.5f * l) + m;
      z0buf[hpack_idx(r, j)] = (_Float16)z;
    }
  }
}

// ------------------------------------------- decoder + fused output heads

__global__ __launch_bounds__(256) void dec_lstm_kernel(
    const float* __restrict__ bvec,
    const _Float16* __restrict__ gwhh,   // dec W_hh frags (step 0 only)
    const _Float16* __restrict__ gwsum,  // dec (W_ih + W_hh) frags
    const _Float16* __restrict__ gwcat,  // [W_mx; W_lx] frags (128 rows)
    const float* __restrict__ bmx, const float* __restrict__ blx,
    _Float16* __restrict__ hbuf, unsigned* __restrict__ bar,
    float* __restrict__ omu, float* __restrict__ ols) {
  extern __shared__ _Float16 lds[];
  _Float16* lwsum = lds;                   // 16 * 8 * 512
  _Float16* lwcat = lds + 16 * 8 * 512;    // 2 * 8 * 512
  const int tid = threadIdx.x, wg = blockIdx.x;
  const int lane = tid & 31, wave = tid >> 5;

#if HAVE_TDM
  if (wave == 0) {
    for (int u = 0; u < 16; ++u) {
      int g = u >> 2, tt = u & 3, nt = g * 16 + wg * 4 + tt;
      tdm_load_1d((unsigned)(uintptr_t)(void*)(lwsum + u * 8 * 512),
                  (const void*)(gwsum + (size_t)nt * 8 * 512), 8 * 512 * 2);
    }
    for (int u = 0; u < 2; ++u) {
      int ct = wg * 2 + u;
      tdm_load_1d((unsigned)(uintptr_t)(void*)(lwcat + u * 8 * 512),
                  (const void*)(gwcat + (size_t)ct * 8 * 512), 8 * 512 * 2);
    }
    __builtin_amdgcn_s_wait_tensorcnt(0);
  }
#else
  for (int u = 0; u < 16; ++u) {
    int g = u >> 2, tt = u & 3, nt = g * 16 + wg * 4 + tt;
    const uint4* s = (const uint4*)(gwsum + (size_t)nt * 8 * 512);
    uint4* d = (uint4*)(lwsum + u * 8 * 512);
    for (int i = tid; i < 512; i += 256) d[i] = s[i];
  }
  for (int u = 0; u < 2; ++u) {
    int ct = wg * 2 + u;
    const uint4* s = (const uint4*)(gwcat + (size_t)ct * 8 * 512);
    uint4* d = (uint4*)(lwcat + u * 8 * 512);
    for (int i = tid; i < 512; i += 256) d[i] = s[i];
  }
#endif
  __syncthreads();

  const int hct = wave >> 1;
  const int mt0 = (wave & 1) * 4;
  const int n = lane & 15;
  const int jcol = wg * 64 + hct * 16 + n;

  float bg[4];
#pragma unroll
  for (int g = 0; g < 4; ++g) bg[g] = bvec[g * HDIM + jcol];

  // Head constants: this wave owns cat col-tile cth, rows 2*(wave&3)..+1
  const int cth = wave >> 2;
  const int cc = wg * 32 + cth * 16 + n;  // 0..63: mu_xhat, 64..127: logsig
  const float bhv = (cc < 64) ? bmx[cc] : blx[cc - 64];
  const int fcol = (cc < 64) ? cc : cc - 64;

  v8f cst[4];
#pragma unroll
  for (int q = 0; q < 4; ++q) cst[q] = splat8(0.0f);

#pragma unroll 1
  for (int t = 0; t < TSTEPS; ++t) {
    const _Float16* hsrc = hbuf + (size_t)(t & 1) * 32768;
    _Float16*       hdst = hbuf + (size_t)((t + 1) & 1) * 32768;
#pragma unroll 1
    for (int q = 0; q < 4; ++q) {
      int mt = mt0 + q;
      v16h ah[8];
#pragma unroll
      for (int kc = 0; kc < 8; ++kc)
        ah[kc] = *(const v16h*)(hsrc + ((mt * 8 + kc) * 32 + lane) * 16);
      v8f acc[4];
#pragma unroll
      for (int g = 0; g < 4; ++g) acc[g] = splat8(bg[g]);
#pragma unroll
      for (int kc = 0; kc < 8; ++kc) {
#pragma unroll
        for (int g = 0; g < 4; ++g) {
          v16h bf;
          if (t == 0) {  // y0 == 0: gates = z0 @ W_hh.T + b
            int nt = g * 16 + wg * 4 + hct;
            bf = *(const v16h*)(gwhh + ((size_t)(nt * 8 + kc) * 32 + lane) * 16);
          } else {       // y == h: gates = h @ (W_ih + W_hh).T + b
            int u = g * 4 + hct;
            bf = *(const v16h*)(lwsum + ((u * 8 + kc) * 32 + lane) * 16);
          }
          acc[g] = wmma16(ah[kc], bf, acc[g]);
        }
      }
#pragma unroll
      for (int k = 0; k < 8; ++k) {
        float iv = sigm(acc[0][k]);
        float fv = sigm(acc[1][k]);
        float gv = tanh_f(acc[2][k]);
        float ov = sigm(acc[3][k]);
        float cn = fv * cst[q][k] + iv * gv;
        cst[q][k] = cn;
        float hn = ov * tanh_f(cn);
        int r = mt * 16 + k + ((lane >> 4) << 3);
        hdst[hpack_idx(r, jcol)] = (_Float16)hn;
      }
    }
    grid_barrier(bar, (unsigned)(NWG * (t + 1)));

    // Fused output heads on h_t (now complete in hdst)
    float* op = (cc < 64) ? omu : ols;
#pragma unroll 1
    for (int q = 0; q < 2; ++q) {
      int mth = (wave & 3) * 2 + q;
      v8f acc = splat8(bhv);
#pragma unroll
      for (int kc = 0; kc < 8; ++kc) {
        v16h a = *(const v16h*)(hdst + ((mth * 8 + kc) * 32 + lane) * 16);
        v16h bf = *(const v16h*)(lwcat + ((cth * 8 + kc) * 32 + lane) * 16);
        acc = wmma16(a, bf, acc);
      }
#pragma unroll
      for (int k = 0; k < 8; ++k) {
        int r = mth * 16 + k + ((lane >> 4) << 3);
        op[((size_t)r * TSTEPS + t) * FDIM + fcol] = acc[k];
      }
    }
  }
}

// ---------------------------------------------------------------- launch

extern "C" void kernel_launch(void* const* d_in, const int* in_sizes, int n_in,
                              void* d_out, int out_size, void* d_ws, size_t ws_size,
                              hipStream_t stream) {
  (void)in_sizes; (void)n_in; (void)out_size; (void)ws_size;
  const float* x    = (const float*)d_in[0];
  const float* eps  = (const float*)d_in[1];
  const float* eWih = (const float*)d_in[2];
  const float* eWhh = (const float*)d_in[3];
  const float* eb   = (const float*)d_in[4];
  const float* Wmp  = (const float*)d_in[5];
  const float* bmp  = (const float*)d_in[6];
  const float* Wlp  = (const float*)d_in[7];
  const float* blp  = (const float*)d_in[8];
  const float* dWih = (const float*)d_in[9];
  const float* dWhh = (const float*)d_in[10];
  const float* db   = (const float*)d_in[11];
  const float* Wmx  = (const float*)d_in[12];
  const float* bmx  = (const float*)d_in[13];
  const float* Wlx  = (const float*)d_in[14];
  const float* blx  = (const float*)d_in[15];

  char* ws = (char*)d_ws;
  _Float16* f_eWih = (_Float16*)(ws + 0);        // 131072 B
  _Float16* f_eWhh = (_Float16*)(ws + 131072);   // 524288 B
  _Float16* f_Wmp  = (_Float16*)(ws + 655360);   // 131072 B
  _Float16* f_Wlp  = (_Float16*)(ws + 786432);   // 131072 B
  _Float16* f_dWhh = (_Float16*)(ws + 917504);   // 524288 B
  _Float16* f_dWs  = (_Float16*)(ws + 1441792);  // 524288 B
  _Float16* f_Wcat = (_Float16*)(ws + 1966080);  // 65536 B
  _Float16* encH   = (_Float16*)(ws + 2031616);  // 131072 B (2 buffers)
  _Float16* decH   = (_Float16*)(ws + 2162688);  // 131072 B (2 buffers)
  unsigned* bars   = (unsigned*)(ws + 2293760);  // 256 B

  // Zero h ping-pong buffers + barrier counters (contiguous region)
  zero_kernel<<<(65600 + 255) / 256, 256, 0, stream>>>((float*)encH, 65600);

  // Pack all weights into WMMA B-fragment layout (f16)
  pack_bfrags<<<16, 256, 0, stream>>>(eWih, nullptr, 1024, 64,  f_eWih);
  pack_bfrags<<<64, 256, 0, stream>>>(eWhh, nullptr, 1024, 256, f_eWhh);
  pack_bfrags<<<16, 256, 0, stream>>>(Wmp,  nullptr, 256,  256, f_Wmp);
  pack_bfrags<<<16, 256, 0, stream>>>(Wlp,  nullptr, 256,  256, f_Wlp);
  pack_bfrags<<<64, 256, 0, stream>>>(dWhh, nullptr, 1024, 256, f_dWhh);
  pack_bfrags<<<64, 256, 0, stream>>>(dWih, dWhh,    1024, 256, f_dWs);   // folded
  pack_bfrags<<<4,  256, 0, stream>>>(Wmx,  nullptr, 64,   256, f_Wcat);
  pack_bfrags<<<4,  256, 0, stream>>>(Wlx,  nullptr, 64,   256, f_Wcat + 16384);

  // Encoder scan (persistent, 4 WGs, 160 KB LDS each)
  enc_lstm_kernel<<<NWG, 256, 163840, stream>>>(x, eb, f_eWih, f_eWhh, encH, bars + 0);

  float* omu_p = (float*)d_out;            // mu_post [1,128,256]
  float* ols_p = omu_p + 32768;            // logsigma2_post
  posterior_kernel<<<1, 256, 0, stream>>>(eps, bmp, blp, f_Wmp, f_Wlp,
                                          encH /* h_T lands in buffer 0 */,
                                          decH /* z0 -> dec buffer 0 */,
                                          omu_p, ols_p);

  float* omu_x = omu_p + 65536;            // mu_xhat [128,1024,64]
  float* ols_x = omu_x + 8388608;          // logsigma2_xhat
  dec_lstm_kernel<<<NWG, 256, 147456, stream>>>(db, f_dWhh, f_dWs, f_Wcat, bmx, blx,
                                                decH, bars + 16, omu_x, ols_x);
}
